// DiffS6_11209864642878
// MI455X (gfx1250) — matrix operand order repaired and checked by reading, not compiled, not run on hardware
//
#include <hip/hip_runtime.h>
#include <hip/hip_bf16.h>

#define D_MODEL 1024
#define D_INNER 2048
#define D_STATE 16
#define DT_RANK 64
#define SEQLEN  2048
#define PROJ    (DT_RANK + 2 * D_STATE)   // 96
#define E2      (2 * D_INNER)             // 4096
#define CHUNK   64
#define NCH     (SEQLEN / CHUNK)          // 32

typedef __attribute__((ext_vector_type(16))) __bf16       v16bf;
typedef __attribute__((ext_vector_type(8)))  float        v8f;
typedef __attribute__((ext_vector_type(8)))  unsigned int v8u;

__device__ __forceinline__ unsigned short f2bf(float f) {
  unsigned int u = __builtin_bit_cast(unsigned int, f);
  u += 0x7FFFu + ((u >> 16) & 1u);           // round-to-nearest-even
  return (unsigned short)(u >> 16);
}
__device__ __forceinline__ float bf2f(unsigned short h) {
  unsigned int u = ((unsigned int)h) << 16;
  return __builtin_bit_cast(float, u);
}

__global__ void k_cvt_bf16(const float* __restrict__ in,
                           unsigned short* __restrict__ out, int n) {
  int i = blockIdx.x * blockDim.x + threadIdx.x;
  if (i < n) out[i] = f2bf(in[i]);
}

// ---- WMMA fragment loaders (layouts per ISA 7.12.2, 16-bit A 16x32 / B 32x16) ----
__device__ __forceinline__ void loadA_frag(const unsigned short* arow, int k, int akb, v8u& au) {
#pragma unroll
  for (int v = 0; v < 8; ++v) {
    const int k0 = ((v & 4) ? 16 : 0) + akb + 2 * (v & 3);
    au[v] = *(const unsigned int*)(arow + k + k0);
  }
}
__device__ __forceinline__ void loadB_frag(const unsigned short* bp, int k, v8u& bu) {
#pragma unroll
  for (int v = 0; v < 8; ++v) bu[v] = *(const unsigned int*)(bp + k + ((v & 4) ? 16 : 0) + 2 * (v & 3));
}

// C (M x N) = A (M x K, bf16, lda) * B^T (B: N x K, bf16, ldb).
// Wave tile = 16(M) x 64(N): 1 A fragment reused over 4 B fragments, 4 accumulators.
// K-loop unrolled by 2 with ping-pong fragment buffers (no register rotation copies).
// Requires K % 64 == 0 (true for all stages here). Block = 4 waves stacked in M.
__global__ __launch_bounds__(128)
void k_gemm_nt(const unsigned short* __restrict__ A, int lda,
               const unsigned short* __restrict__ B, int ldb,
               float* __restrict__ Cf, int ldc,
               unsigned short* __restrict__ CbT, int ldct,
               int M, int N, int K) {
  const int wave = threadIdx.x >> 5;
  const int lane = threadIdx.x & 31;
  const int m0 = blockIdx.y * 64 + wave * 16;
  const int n0 = blockIdx.x * 64;
  if (m0 >= M) return;                       // wave-uniform; branched over (no WMMA w/ EXEC=0)

  const int am  = m0 + (lane & 15);
  const int akb = (lane >> 4) * 8;
  const int bn  = (lane & 15);
  const int bkb = (lane >> 4) * 8;

  const unsigned short* arow = A + (long)am * lda;
  const unsigned short* bp[4];
#pragma unroll
  for (int j = 0; j < 4; ++j)
    bp[j] = B + (long)(n0 + j * 16 + bn) * ldb + bkb;

  v8f acc[4];
#pragma unroll
  for (int j = 0; j < 4; ++j) acc[j] = (v8f){0.f,0.f,0.f,0.f,0.f,0.f,0.f,0.f};

  v8u a0, b0[4], a1, b1[4];
  loadA_frag(arow, 0, akb, a0);
#pragma unroll
  for (int j = 0; j < 4; ++j) loadB_frag(bp[j], 0, b0[j]);

  for (int k = 0; k < K; k += 64) {
    // preload k+32 (always in range since K % 64 == 0)
    loadA_frag(arow, k + 32, akb, a1);
#pragma unroll
    for (int j = 0; j < 4; ++j) loadB_frag(bp[j], k + 32, b1[j]);
    __builtin_prefetch(arow + k + 512, 0, 3);

    {
      const v16bf av = __builtin_bit_cast(v16bf, a0);
#pragma unroll
      for (int j = 0; j < 4; ++j)
        acc[j] = __builtin_amdgcn_wmma_f32_16x16x32_bf16(
            false, av, false, __builtin_bit_cast(v16bf, b0[j]),
            (short)0, acc[j], false, false);
    }

    if (k + 64 < K) {                        // preload k+64 into buffer 0
      loadA_frag(arow, k + 64, akb, a0);
#pragma unroll
      for (int j = 0; j < 4; ++j) loadB_frag(bp[j], k + 64, b0[j]);
    }

    {
      const v16bf av = __builtin_bit_cast(v16bf, a1);
#pragma unroll
      for (int j = 0; j < 4; ++j)
        acc[j] = __builtin_amdgcn_wmma_f32_16x16x32_bf16(
            false, av, false, __builtin_bit_cast(v16bf, b1[j]),
            (short)0, acc[j], false, false);
    }
  }

  // C/D layout: VGPR r -> row m0 + r + 8*(lane>=16)
  const int row0 = m0 + 8 * (lane >> 4);
#pragma unroll
  for (int j = 0; j < 4; ++j) {
    const int col = n0 + j * 16 + (lane & 15);
#pragma unroll
    for (int r = 0; r < 8; ++r) {
      const int row = row0 + r;
      if (Cf)  Cf[(long)row * ldc + col]   = acc[j][r];
      if (CbT) CbT[(long)col * ldct + row] = f2bf(acc[j][r]);
    }
  }
}

// Depthwise causal conv (w=4) + bias + SiLU -> x_ld (l,d) bf16; also z -> zT (l,d) bf16.
__global__ __launch_bounds__(256)
void k_conv_silu(const float* __restrict__ xz, const float* __restrict__ cw,
                 const float* __restrict__ cb,
                 unsigned short* __restrict__ x_ld, unsigned short* __restrict__ zT) {
  int l = blockIdx.x * blockDim.x + threadIdx.x;
  int d = blockIdx.y;
  if (l >= SEQLEN) return;
  const float* xr = xz + (long)d * SEQLEN;
  float a = cb[d];
  if (l >= 3) a += cw[d * 4 + 0] * xr[l - 3];
  if (l >= 2) a += cw[d * 4 + 1] * xr[l - 2];
  if (l >= 1) a += cw[d * 4 + 2] * xr[l - 1];
  a += cw[d * 4 + 3] * xr[l];
  float s = a / (1.0f + __expf(-a));          // SiLU
  x_ld[(long)l * D_INNER + d] = f2bf(s);
  float zv = xz[(long)(D_INNER + d) * SEQLEN + l];
  zT[(long)l * D_INNER + d] = f2bf(zv);
}

__device__ __forceinline__ float softplus_f(float x) {
  return (x > 20.f) ? x : __logf(1.0f + __expf(x));
}

// ---- Chunked selective scan: h_t = a_t h_{t-1} + b_t, a_t = exp(dt*A), b_t = dt*u*B_t ----
// Pass 1: per (d, chunk) compute P = prod a_t and S = local scan end-state (h0 = 0).
__global__ __launch_bounds__(256)
void k_scan_pass1(const unsigned short* __restrict__ x_ld,
                  const unsigned short* __restrict__ dtT1, const unsigned short* __restrict__ dtT2,
                  const unsigned short* __restrict__ dblT1, const unsigned short* __restrict__ dblT2,
                  const float* __restrict__ A1l, const float* __restrict__ A2l,
                  const float* __restrict__ dtb1, const float* __restrict__ dtb2,
                  float* __restrict__ P1, float* __restrict__ S1,
                  float* __restrict__ P2, float* __restrict__ S2) {
  const int d = blockIdx.y * blockDim.x + threadIdx.x;
  const int c = blockIdx.x;
  float A1[D_STATE], A2[D_STATE], p1[D_STATE], s1[D_STATE], p2[D_STATE], s2[D_STATE];
#pragma unroll
  for (int n = 0; n < D_STATE; ++n) {
    A1[n] = -__expf(A1l[(long)d * D_STATE + n]);
    A2[n] = -__expf(A2l[(long)d * D_STATE + n]);
    p1[n] = 1.f; s1[n] = 0.f; p2[n] = 1.f; s2[n] = 0.f;
  }
  const float b1 = dtb1[d], b2 = dtb2[d];
  const int l0 = c * CHUNK;
  for (int t = 0; t < CHUNK; ++t) {
    const int l = l0 + t;
    const float u = bf2f(x_ld[(long)l * D_INNER + d]);

    float dt = softplus_f(bf2f(dtT1[(long)l * D_INNER + d]) + b1);
    const unsigned short* bc = dblT1 + (long)l * PROJ + DT_RANK;
    float du = dt * u;
#pragma unroll
    for (int n = 0; n < D_STATE; ++n) {
      const float dA = __expf(dt * A1[n]);
      p1[n] *= dA;
      s1[n] = dA * s1[n] + du * bf2f(bc[n]);
    }
    dt = softplus_f(bf2f(dtT2[(long)l * D_INNER + d]) + b2);
    bc = dblT2 + (long)l * PROJ + DT_RANK;
    du = dt * u;
#pragma unroll
    for (int n = 0; n < D_STATE; ++n) {
      const float dA = __expf(dt * A2[n]);
      p2[n] *= dA;
      s2[n] = dA * s2[n] + du * bf2f(bc[n]);
    }
  }
  const long base = ((long)c * D_INNER + d) * D_STATE;
#pragma unroll
  for (int n = 0; n < D_STATE; ++n) {
    P1[base + n] = p1[n]; S1[base + n] = s1[n];
    P2[base + n] = p2[n]; S2[base + n] = s2[n];
  }
}

// Pass 2: inter-chunk recurrence per (d,n): Hinit[c] = S[c-1] + P[c-1]*Hinit[c-1].
__global__ __launch_bounds__(256)
void k_scan_pass2(const float* __restrict__ P1, const float* __restrict__ S1,
                  const float* __restrict__ P2, const float* __restrict__ S2,
                  float* __restrict__ Hi1, float* __restrict__ Hi2) {
  const int i = blockIdx.x * blockDim.x + threadIdx.x;   // i = d*16 + n
  if (i >= D_INNER * D_STATE) return;
  float h1 = 0.f, h2 = 0.f;
  for (int c = 0; c < NCH; ++c) {
    const long idx = (long)c * (D_INNER * D_STATE) + i;
    Hi1[idx] = h1; Hi2[idx] = h2;
    h1 = S1[idx] + P1[idx] * h1;
    h2 = S2[idx] + P2[idx] * h2;
  }
}

// Pass 3: replay each chunk with correct initial state; emit y = (scan + D*u)*silu(z), diff.
__global__ __launch_bounds__(256)
void k_scan_pass3(const unsigned short* __restrict__ x_ld, const unsigned short* __restrict__ zT,
                  const unsigned short* __restrict__ dtT1, const unsigned short* __restrict__ dtT2,
                  const unsigned short* __restrict__ dblT1, const unsigned short* __restrict__ dblT2,
                  const float* __restrict__ A1l, const float* __restrict__ A2l,
                  const float* __restrict__ D1, const float* __restrict__ D2,
                  const float* __restrict__ dtb1, const float* __restrict__ dtb2,
                  const float* __restrict__ Hi1, const float* __restrict__ Hi2,
                  unsigned short* __restrict__ yT) {
  const int d = blockIdx.y * blockDim.x + threadIdx.x;
  const int c = blockIdx.x;
  float A1[D_STATE], A2[D_STATE], h1[D_STATE], h2[D_STATE];
  const long base = ((long)c * D_INNER + d) * D_STATE;
#pragma unroll
  for (int n = 0; n < D_STATE; ++n) {
    A1[n] = -__expf(A1l[(long)d * D_STATE + n]);
    A2[n] = -__expf(A2l[(long)d * D_STATE + n]);
    h1[n] = Hi1[base + n];
    h2[n] = Hi2[base + n];
  }
  const float Dv1 = D1[d], Dv2 = D2[d], b1 = dtb1[d], b2 = dtb2[d];
  const int l0 = c * CHUNK;
  for (int t = 0; t < CHUNK; ++t) {
    const int l = l0 + t;
    const float u  = bf2f(x_ld[(long)l * D_INNER + d]);
    const float zv = bf2f(zT[(long)l * D_INNER + d]);
    const float sz = zv / (1.0f + __expf(-zv));

    float dt = softplus_f(bf2f(dtT1[(long)l * D_INNER + d]) + b1);
    const unsigned short* bc = dblT1 + (long)l * PROJ + DT_RANK;
    float du = dt * u, y1 = 0.f;
#pragma unroll
    for (int n = 0; n < D_STATE; ++n) {
      const float dA = __expf(dt * A1[n]);
      h1[n] = dA * h1[n] + du * bf2f(bc[n]);
      y1 += h1[n] * bf2f(bc[D_STATE + n]);
    }
    y1 = (y1 + Dv1 * u) * sz;

    dt = softplus_f(bf2f(dtT2[(long)l * D_INNER + d]) + b2);
    bc = dblT2 + (long)l * PROJ + DT_RANK;
    du = dt * u;
    float y2 = 0.f;
#pragma unroll
    for (int n = 0; n < D_STATE; ++n) {
      const float dA = __expf(dt * A2[n]);
      h2[n] = dA * h2[n] + du * bf2f(bc[n]);
      y2 += h2[n] * bf2f(bc[D_STATE + n]);
    }
    y2 = (y2 + Dv2 * u) * sz;

    yT[(long)l * D_INNER + d] = f2bf(y1 - y2);
  }
}

extern "C" void kernel_launch(void* const* d_in, const int* in_sizes, int n_in,
                              void* d_out, int out_size, void* d_ws, size_t ws_size,
                              hipStream_t stream) {
  const float* hidden = (const float*)d_in[0];
  const float* inw    = (const float*)d_in[1];
  const float* conv_w = (const float*)d_in[2];
  const float* conv_b = (const float*)d_in[3];
  const float* x1pw   = (const float*)d_in[4];
  const float* dt1w   = (const float*)d_in[5];
  const float* dt1b   = (const float*)d_in[6];
  const float* A1l    = (const float*)d_in[7];
  const float* D1     = (const float*)d_in[8];
  const float* x2pw   = (const float*)d_in[9];
  const float* dt2w   = (const float*)d_in[10];
  const float* dt2b   = (const float*)d_in[11];
  const float* A2l    = (const float*)d_in[12];
  const float* D2     = (const float*)d_in[13];
  const float* outw   = (const float*)d_in[14];
  float* out = (float*)d_out;
  (void)in_sizes; (void)n_in; (void)out_size; (void)ws_size;

  char* base = (char*)d_ws;
  size_t off = 0;
  auto alloc = [&](size_t bytes) -> char* {
    char* p = base + off;
    off += (bytes + 255) & ~(size_t)255;
    return p;
  };
  unsigned short* hid_bf  = (unsigned short*)alloc((size_t)SEQLEN * D_MODEL * 2);
  unsigned short* inw_bf  = (unsigned short*)alloc((size_t)E2 * D_MODEL * 2);
  unsigned short* x1pw_bf = (unsigned short*)alloc((size_t)PROJ * D_INNER * 2);
  unsigned short* x2pw_bf = (unsigned short*)alloc((size_t)PROJ * D_INNER * 2);
  unsigned short* dt1w_bf = (unsigned short*)alloc((size_t)D_INNER * DT_RANK * 2);
  unsigned short* dt2w_bf = (unsigned short*)alloc((size_t)D_INNER * DT_RANK * 2);
  unsigned short* outw_bf = (unsigned short*)alloc((size_t)D_MODEL * D_INNER * 2);
  float*          xz      = (float*)alloc((size_t)E2 * SEQLEN * 4);
  unsigned short* x_ld    = (unsigned short*)alloc((size_t)SEQLEN * D_INNER * 2);
  unsigned short* zT      = (unsigned short*)alloc((size_t)SEQLEN * D_INNER * 2);
  unsigned short* dblT1   = (unsigned short*)alloc((size_t)SEQLEN * PROJ * 2);
  unsigned short* dblT2   = (unsigned short*)alloc((size_t)SEQLEN * PROJ * 2);
  unsigned short* dtT1    = (unsigned short*)alloc((size_t)SEQLEN * D_INNER * 2);
  unsigned short* dtT2    = (unsigned short*)alloc((size_t)SEQLEN * D_INNER * 2);
  unsigned short* yT      = (unsigned short*)alloc((size_t)SEQLEN * D_INNER * 2);
  float* P1  = (float*)alloc((size_t)NCH * D_INNER * D_STATE * 4);
  float* S1  = (float*)alloc((size_t)NCH * D_INNER * D_STATE * 4);
  float* P2  = (float*)alloc((size_t)NCH * D_INNER * D_STATE * 4);
  float* S2  = (float*)alloc((size_t)NCH * D_INNER * D_STATE * 4);
  float* Hi1 = (float*)alloc((size_t)NCH * D_INNER * D_STATE * 4);
  float* Hi2 = (float*)alloc((size_t)NCH * D_INNER * D_STATE * 4);

  auto cvt = [&](const float* s, unsigned short* dst, long n) {
    k_cvt_bf16<<<(unsigned)((n + 255) / 256), 256, 0, stream>>>(s, dst, (int)n);
  };
  cvt(hidden, hid_bf, (long)SEQLEN * D_MODEL);
  cvt(inw,    inw_bf, (long)E2 * D_MODEL);
  cvt(x1pw,   x1pw_bf, (long)PROJ * D_INNER);
  cvt(x2pw,   x2pw_bf, (long)PROJ * D_INNER);
  cvt(dt1w,   dt1w_bf, (long)D_INNER * DT_RANK);
  cvt(dt2w,   dt2w_bf, (long)D_INNER * DT_RANK);
  cvt(outw,   outw_bf, (long)D_MODEL * D_INNER);

  // 1) in_proj: xz (E2 x L) = inw * hidden^T
  {
    dim3 g(SEQLEN / 64, E2 / 64);
    k_gemm_nt<<<g, 128, 0, stream>>>(inw_bf, D_MODEL, hid_bf, D_MODEL,
                                     xz, SEQLEN, nullptr, 0, E2, SEQLEN, D_MODEL);
  }
  // 2) causal conv + SiLU -> x_ld, z -> zT
  {
    dim3 g(SEQLEN / 256, D_INNER);
    k_conv_silu<<<g, 256, 0, stream>>>(xz, conv_w, conv_b, x_ld, zT);
  }
  // 3) x_proj branches -> dblT (L x 96, bf16); M=96 tiles to 2 blocks (idle waves return)
  {
    dim3 g(SEQLEN / 64, (PROJ + 63) / 64);
    k_gemm_nt<<<g, 128, 0, stream>>>(x1pw_bf, D_INNER, x_ld, D_INNER,
                                     nullptr, 0, dblT1, PROJ, PROJ, SEQLEN, D_INNER);
    k_gemm_nt<<<g, 128, 0, stream>>>(x2pw_bf, D_INNER, x_ld, D_INNER,
                                     nullptr, 0, dblT2, PROJ, PROJ, SEQLEN, D_INNER);
  }
  // 4) dt projections -> dtT (L x D_INNER, bf16); B = first 64 cols of dblT (ldb=96)
  {
    dim3 g(SEQLEN / 64, D_INNER / 64);
    k_gemm_nt<<<g, 128, 0, stream>>>(dt1w_bf, DT_RANK, dblT1, PROJ,
                                     nullptr, 0, dtT1, D_INNER, D_INNER, SEQLEN, DT_RANK);
    k_gemm_nt<<<g, 128, 0, stream>>>(dt2w_bf, DT_RANK, dblT2, PROJ,
                                     nullptr, 0, dtT2, D_INNER, D_INNER, SEQLEN, DT_RANK);
  }
  // 5) chunked dual selective scan -> yT (L x D_INNER, bf16)
  {
    dim3 g1(NCH, D_INNER / 256);
    k_scan_pass1<<<g1, 256, 0, stream>>>(x_ld, dtT1, dtT2, dblT1, dblT2,
                                         A1l, A2l, dt1b, dt2b, P1, S1, P2, S2);
    k_scan_pass2<<<(D_INNER * D_STATE) / 256, 256, 0, stream>>>(P1, S1, P2, S2, Hi1, Hi2);
    k_scan_pass3<<<g1, 256, 0, stream>>>(x_ld, zT, dtT1, dtT2, dblT1, dblT2,
                                         A1l, A2l, D1, D2, dt1b, dt2b, Hi1, Hi2, yT);
  }
  // 6) out_proj: out (L x Dm) = yT (L x Di) * outw^T -> fp32 directly to d_out
  {
    dim3 g(D_MODEL / 64, SEQLEN / 64);
    k_gemm_nt<<<g, 128, 0, stream>>>(yT, D_INNER, outw_bf, D_INNER,
                                     out, D_MODEL, nullptr, 0, SEQLEN, D_MODEL, D_INNER);
  }
}
